// AttentionOnlyHyperedgeGenerator_76562087019170
// MI455X (gfx1250) — matrix operand order repaired and checked by reading, not compiled, not run on hardware
//
#include <hip/hip_runtime.h>
#include <math.h>

// ---------------------------------------------------------------------------
// CDNA5 (gfx1250) WMMA types
// ---------------------------------------------------------------------------
typedef __attribute__((ext_vector_type(16))) _Float16     v16h;
typedef __attribute__((ext_vector_type(8)))  float        v8f;
typedef __attribute__((ext_vector_type(4)))  unsigned int v4u;

union Frag { v16h h; v4u u[2]; };

// A-fragment: 16x32 f16, row-major source [.., lda].
// Per ISA 7.12.2: lane L (m=L&15, half=L>>4) holds
//   elements 0..7  -> K = kb + 8*half + (0..7)
//   elements 8..15 -> K = kb + 16 + 8*half + (0..7)
static __device__ __forceinline__ v16h load_a(const _Float16* A, long row0, int lda,
                                              int kb, int m, int half) {
  Frag f;
  const _Float16* p = A + (row0 + m) * (long)lda + kb + half * 8;
  f.u[0] = *(const v4u*)p;
  f.u[1] = *(const v4u*)(p + 16);
  return f.h;
}

// B-fragment: 32x16 f16 from Bt stored row-major [N,K] (i.e. B transposed).
// Lane L: n = n0 + (L&15); elements 0..7 -> K = kb + 8*(L>>4) + e, 8..15 -> K = kb + 16 + 8*(L>>4) + (e-8).
static __device__ __forceinline__ v16h load_b(const _Float16* Bt, long n0, int K,
                                              int kb, int m, int half) {
  Frag f;
  const _Float16* p = Bt + (n0 + m) * (long)K + kb + half * 8;
  f.u[0] = *(const v4u*)p;
  f.u[1] = *(const v4u*)(p + 16);
  return f.h;
}

// C/D layout: element v of lane L -> row (v + 8*(L>>4)), col (L&15).
static __device__ __forceinline__ void store_tile(v8f acc, long row0, long n, int half,
                                                  const float* bias, float* Cf,
                                                  _Float16* Ch, int ldc, int relu) {
  float bv = bias ? bias[n] : 0.0f;
#pragma unroll
  for (int v = 0; v < 8; ++v) {
    float x = acc[v] + bv;
    if (relu) x = fmaxf(x, 0.0f);
    long off = (row0 + v + 8 * half) * (long)ldc + n;
    if (Cf) Cf[off] = x;
    if (Ch) Ch[off] = (_Float16)x;
  }
}

// Generic f16 WMMA GEMM: C[M,N] = act(A[M,K] @ Bt[N,K]^T + bias).
// One wave per block; each wave computes a 32x64 tile (2 row-tiles x 4 N-tiles):
// per k-step 12 b128 loads feed 8 v_wmma (B fragments reused across row-tiles),
// and 8 independent accumulator chains hide the WMMA->WMMA RAW latency.
// grid = (N/64, M/32). Operands are small enough to live in MI455X's 192MB L2,
// so no LDS staging is required.
__global__ void __launch_bounds__(32) gemm_wmma(const _Float16* __restrict__ A,
                                                const _Float16* __restrict__ Bt,
                                                const float* __restrict__ bias,
                                                float* __restrict__ Cf,
                                                _Float16* __restrict__ Ch,
                                                int K, int lda, int ldc, int relu) {
  const int lane = threadIdx.x;
  const int m = lane & 15, half = lane >> 4;
  const long row0 = (long)blockIdx.y * 32;
  const long n0   = (long)blockIdx.x * 64;

  v8f acc[2][4];
#pragma unroll
  for (int r = 0; r < 2; ++r)
#pragma unroll
    for (int t = 0; t < 4; ++t)
#pragma unroll
      for (int v = 0; v < 8; ++v) acc[r][t][v] = 0.f;

  for (int kb = 0; kb < K; kb += 32) {
    v16h a0 = load_a(A, row0,      lda, kb, m, half);
    v16h a1 = load_a(A, row0 + 16, lda, kb, m, half);
#pragma unroll
    for (int t = 0; t < 4; ++t) {
      v16h b = load_b(Bt, n0 + 16 * t, K, kb, m, half);
      acc[0][t] = __builtin_amdgcn_wmma_f32_16x16x32_f16(false, a0, false, b, (short)0,
                                                         acc[0][t], false, false);
      acc[1][t] = __builtin_amdgcn_wmma_f32_16x16x32_f16(false, a1, false, b, (short)0,
                                                         acc[1][t], false, false);
    }
  }
#pragma unroll
  for (int r = 0; r < 2; ++r)
#pragma unroll
    for (int t = 0; t < 4; ++t)
      store_tile(acc[r][t], row0 + 16 * r, n0 + 16 * t + m, half, bias, Cf, Ch, ldc, relu);
}

// ---------------------------------------------------------------------------
// Converts
// ---------------------------------------------------------------------------
__global__ void cvt_f16(const float* __restrict__ s, _Float16* __restrict__ d, long n) {
  long i = (long)blockIdx.x * blockDim.x + threadIdx.x;
  long str = (long)gridDim.x * blockDim.x;
  for (; i < n; i += str) d[i] = (_Float16)s[i];
}

__global__ void cvt_tr(const float* __restrict__ W, _Float16* __restrict__ Wt, int K, int N) {
  long total = (long)K * N;
  long i = (long)blockIdx.x * blockDim.x + threadIdx.x;
  long str = (long)gridDim.x * blockDim.x;
  for (; i < total; i += str) {
    long k = i / N, n = i % N;
    Wt[n * (long)K + k] = (_Float16)W[i];
  }
}

// ---------------------------------------------------------------------------
// LayerNorm over rows of [3B,256] (modality = row % 3), output f16
// ---------------------------------------------------------------------------
__global__ void __launch_bounds__(256) ln_kernel(const float* __restrict__ pre,
                                                 _Float16* __restrict__ out,
                                                 const float* g0c, const float* g1c,
                                                 const float* g2c, const float* b0c,
                                                 const float* b1c, const float* b2c) {
  int lane = threadIdx.x & 31;
  long r = (long)blockIdx.x * 8 + (threadIdx.x >> 5);
  int mod = (int)(r % 3);
  const float* g  = mod == 0 ? g0c : (mod == 1 ? g1c : g2c);
  const float* be = mod == 0 ? b0c : (mod == 1 ? b1c : b2c);
  const float* row = pre + r * 256;
  float x[8]; float s = 0.f;
#pragma unroll
  for (int j = 0; j < 8; ++j) { x[j] = row[lane * 8 + j]; s += x[j]; }
  for (int msk = 16; msk >= 1; msk >>= 1) s += __shfl_xor(s, msk, 32);
  float mean = s * (1.0f / 256.0f);
  float vs = 0.f;
#pragma unroll
  for (int j = 0; j < 8; ++j) { float d = x[j] - mean; vs += d * d; }
  for (int msk = 16; msk >= 1; msk >>= 1) vs += __shfl_xor(vs, msk, 32);
  float rstd = rsqrtf(vs * (1.0f / 256.0f) + 1e-5f);
  _Float16* orow = out + r * 256;
#pragma unroll
  for (int j = 0; j < 8; ++j) {
    int c = lane * 8 + j;
    orow[c] = (_Float16)((x[j] - mean) * rstd * g[c] + be[c]);
  }
}

// ---------------------------------------------------------------------------
// Tiny 3-token / 4-head attention. One wave per batch row; lane handles 8
// contiguous dims, lane-groups of 8 == one 64-dim head.
// ---------------------------------------------------------------------------
__global__ void __launch_bounds__(256) attn_kernel(const _Float16* __restrict__ qkv,
                                                   _Float16* __restrict__ o) {
  int lane = threadIdx.x & 31;
  long b = (long)blockIdx.x * 8 + (threadIdx.x >> 5);
  const _Float16* base = qkv + b * 3 * 768;
  int c0 = lane * 8;
  float q[3][8], k[3][8], v[3][8];
#pragma unroll
  for (int t = 0; t < 3; ++t)
#pragma unroll
    for (int j = 0; j < 8; ++j) {
      q[t][j] = (float)base[t * 768 +       c0 + j];
      k[t][j] = (float)base[t * 768 + 256 + c0 + j];
      v[t][j] = (float)base[t * 768 + 512 + c0 + j];
    }
  float s[3][3];
#pragma unroll
  for (int t1 = 0; t1 < 3; ++t1)
#pragma unroll
    for (int t2 = 0; t2 < 3; ++t2) {
      float p = 0.f;
#pragma unroll
      for (int j = 0; j < 8; ++j) p += q[t1][j] * k[t2][j];
      s[t1][t2] = p;
    }
  for (int msk = 1; msk < 8; msk <<= 1)
#pragma unroll
    for (int t1 = 0; t1 < 3; ++t1)
#pragma unroll
      for (int t2 = 0; t2 < 3; ++t2)
        s[t1][t2] += __shfl_xor(s[t1][t2], msk, 32);
  float p[3][3];
#pragma unroll
  for (int t1 = 0; t1 < 3; ++t1) {
    float a0 = s[t1][0] * 0.125f, a1 = s[t1][1] * 0.125f, a2 = s[t1][2] * 0.125f;
    float mx = fmaxf(a0, fmaxf(a1, a2));
    float e0 = expf(a0 - mx), e1 = expf(a1 - mx), e2 = expf(a2 - mx);
    float inv = 1.0f / (e0 + e1 + e2);
    p[t1][0] = e0 * inv; p[t1][1] = e1 * inv; p[t1][2] = e2 * inv;
  }
#pragma unroll
  for (int t1 = 0; t1 < 3; ++t1) {
    _Float16* orow = o + (b * 3 + t1) * 256 + c0;
#pragma unroll
    for (int j = 0; j < 8; ++j) {
      float ov = p[t1][0] * v[0][j] + p[t1][1] * v[1][j] + p[t1][2] * v[2][j];
      orow[j] = (_Float16)ov;
    }
  }
}

// fused = mean over the 3 modality rows, emit f32 + f16 copies
__global__ void __launch_bounds__(256) mean_kernel(const float* __restrict__ oW,
                                                   float* __restrict__ fo,
                                                   _Float16* __restrict__ fh) {
  long i = (long)blockIdx.x * 256 + threadIdx.x;  // over B*256
  long bb = i >> 8; long c = i & 255;
  const float* p = oW + bb * 3 * 256 + c;
  float x = (p[0] + p[256] + p[512]) * (1.0f / 3.0f);
  fo[i] = x; fh[i] = (_Float16)x;
}

// ---------------------------------------------------------------------------
// Row softmax stats + top-8 over raw scores (stored temporarily in H).
// Row cached in 32KB LDS (CDNA5 WGP has 320KB); 8 argmax passes,
// lower-index tie-break to match jax.lax.top_k.
// ---------------------------------------------------------------------------
__global__ void __launch_bounds__(256) softmax_topk(const float* __restrict__ S,
                                                    float* __restrict__ vals,
                                                    int* __restrict__ idxo) {
  __shared__ float row[8192];
  __shared__ float rv[256];
  __shared__ int   ri[256];
  const long i = blockIdx.x;
  const int t = threadIdx.x;
  const float* src = S + i * 8192;
  float lmax = -3.4e38f;
  for (int j = t; j < 8192; j += 256) { float x = src[j]; row[j] = x; lmax = fmaxf(lmax, x); }
  rv[t] = lmax; __syncthreads();
  for (int s2 = 128; s2 > 0; s2 >>= 1) { if (t < s2) rv[t] = fmaxf(rv[t], rv[t + s2]); __syncthreads(); }
  float rmax = rv[0]; __syncthreads();
  float lsum = 0.f;
  for (int j = t; j < 8192; j += 256) lsum += expf(row[j] - rmax);
  rv[t] = lsum; __syncthreads();
  for (int s2 = 128; s2 > 0; s2 >>= 1) { if (t < s2) rv[t] += rv[t + s2]; __syncthreads(); }
  float rsum = rv[0]; __syncthreads();
  for (int kk = 0; kk < 8; ++kk) {
    float bv = -3.4e38f; int bi = 0x7fffffff;
    for (int j = t; j < 8192; j += 256) { float x = row[j]; if (x > bv) { bv = x; bi = j; } }
    rv[t] = bv; ri[t] = bi; __syncthreads();
    for (int s2 = 128; s2 > 0; s2 >>= 1) {
      if (t < s2) {
        if (rv[t + s2] > rv[t] || (rv[t + s2] == rv[t] && ri[t + s2] < ri[t])) {
          rv[t] = rv[t + s2]; ri[t] = ri[t + s2];
        }
      }
      __syncthreads();
    }
    if (t == 0) {
      int bj = ri[0];
      vals[i * 8 + kk] = expf(rv[0] - rmax) / rsum;
      idxo[i * 8 + kk] = bj;
      row[bj] = -3.4e38f;
    }
    __syncthreads();
  }
}

// H = eye(B): vectorized float4 stores, grid-stride
__global__ void init_eye(float* __restrict__ H) {
  long idx = (long)blockIdx.x * blockDim.x + threadIdx.x;
  long stride = (long)gridDim.x * blockDim.x;
  const long total4 = (long)8192 * 8192 / 4;
  for (; idx < total4; idx += stride) {
    long p = idx * 4;
    long rrow = p >> 13;
    long c0 = p & 8191;
    float4 val = make_float4(0.f, 0.f, 0.f, 0.f);
    long d = rrow - c0;
    if (d >= 0 && d < 4) ((float*)&val)[d] = 1.0f;
    *(float4*)(H + p) = val;
  }
}

// Scatter H[idx[i,j], i] = vals[i,j]; column i is written ONLY by row i's
// scatter, so colsum[i] = sum(vals[i]) + (i in idx[i] ? 0 : 1) -> exact fallback.
__global__ void __launch_bounds__(256) scatter_kernel(float* __restrict__ H,
                                                      const float* __restrict__ vals,
                                                      const int* __restrict__ idx) {
  long i = (long)blockIdx.x * 256 + threadIdx.x;
  float s = 0.f; int diag = 0;
#pragma unroll
  for (int j = 0; j < 8; ++j) {
    float v = vals[i * 8 + j]; int r = idx[i * 8 + j];
    H[(long)r * 8192 + i] = v;
    s += v; if (r == (int)i) diag = 1;
  }
  float colsum = s + (diag ? 0.f : 1.0f);
  if (colsum == 0.0f) H[i * 8192 + i] = 1.0f;
}

// edge weights: sigmoid(relu(fused@W1+b1)@W2+b2), clamp min 1e-8. Wave per row.
__global__ void __launch_bounds__(256) edge_weight(const float* __restrict__ fused,
                                                   const float* __restrict__ W1,
                                                   const float* __restrict__ b1,
                                                   const float* __restrict__ W2,
                                                   const float* __restrict__ b2,
                                                   float* __restrict__ ew) {
  int lane = threadIdx.x & 31;
  long bb = (long)blockIdx.x * 8 + (threadIdx.x >> 5);
  const float* f = fused + bb * 256;
  float part = 0.f;
#pragma unroll
  for (int jj = 0; jj < 4; ++jj) {
    int j = lane + 32 * jj;
    float acc = b1[j];
    for (int k2 = 0; k2 < 256; ++k2) acc = fmaf(f[k2], W1[k2 * 128 + j], acc);
    part += fmaxf(acc, 0.f) * W2[j];
  }
  for (int msk = 16; msk >= 1; msk >>= 1) part += __shfl_xor(part, msk, 32);
  if (lane == 0) {
    float e = 1.0f / (1.0f + expf(-(part + b2[0])));
    ew[bb] = fmaxf(e, 1e-8f);
  }
}

// ---------------------------------------------------------------------------
// Launch (ws usage ~90.3 MB with region reuse; raw scores staged in d_out)
// ---------------------------------------------------------------------------
extern "C" void kernel_launch(void* const* d_in, const int* in_sizes, int n_in,
                              void* d_out, int out_size, void* d_ws, size_t ws_size,
                              hipStream_t stream) {
  (void)in_sizes; (void)n_in; (void)out_size; (void)ws_size;
  const float* x0  = (const float*)d_in[0];
  const float* Wp0 = (const float*)d_in[1];
  const float* bp0 = (const float*)d_in[2];
  const float* g0  = (const float*)d_in[3];
  const float* be0 = (const float*)d_in[4];
  const float* x1  = (const float*)d_in[5];
  const float* Wp1 = (const float*)d_in[6];
  const float* bp1 = (const float*)d_in[7];
  const float* g1  = (const float*)d_in[8];
  const float* be1 = (const float*)d_in[9];
  const float* x2  = (const float*)d_in[10];
  const float* Wp2 = (const float*)d_in[11];
  const float* bp2 = (const float*)d_in[12];
  const float* g2  = (const float*)d_in[13];
  const float* be2 = (const float*)d_in[14];
  const float* Win = (const float*)d_in[15];
  const float* binv= (const float*)d_in[16];
  const float* Wout= (const float*)d_in[17];
  const float* bout= (const float*)d_in[18];
  const float* W1  = (const float*)d_in[19];
  const float* b1  = (const float*)d_in[20];
  const float* W2  = (const float*)d_in[21];
  const float* b2  = (const float*)d_in[22];

  const long Bn = 8192;
  _Float16* xb   = (_Float16*)d_ws;               // [B,2304] f16 (reused as qkv f16)
  _Float16* x0b  = xb;
  _Float16* x1b  = x0b + Bn * 512;
  _Float16* x2b  = x1b + Bn * 768;
  _Float16* Wp0t = xb + Bn * 2304;                // transposed f16 weights [N,K]
  _Float16* Wp1t = Wp0t + 256L * 512;
  _Float16* Wp2t = Wp1t + 256L * 768;
  _Float16* Wint = Wp2t + 256L * 1024;
  _Float16* Woutt= Wint + 768L * 256;
  float*    seqpre = (float*)(Woutt + 256L * 256);   // [3B,256] f32 (reused as o@Wout)
  _Float16* seqb   = (_Float16*)(seqpre + 24576L * 256); // [3B,256] f16 (reused as attn out)
  float*    fusedf = (float*)(seqb + 24576L * 256);
  _Float16* fusedh = (_Float16*)(fusedf + Bn * 256);
  float*    tvals  = (float*)(fusedh + Bn * 256);
  int*      tidx   = (int*)(tvals + Bn * 8);
  _Float16* qkvb = xb;     // alias: x f16 dead after proj GEMMs
  _Float16* ob   = seqb;   // alias: seqb dead after qkv GEMM
  float*    oWp  = seqpre; // alias: seqpre dead after LN
  float* H  = (float*)d_out;
  float* ew = H + Bn * Bn;

  // converts
  cvt_f16<<<2048, 256, 0, stream>>>(x0, x0b, Bn * 512);
  cvt_f16<<<2048, 256, 0, stream>>>(x1, x1b, Bn * 768);
  cvt_f16<<<2048, 256, 0, stream>>>(x2, x2b, Bn * 1024);
  cvt_tr<<<512, 256, 0, stream>>>(Wp0, Wp0t, 512, 256);
  cvt_tr<<<512, 256, 0, stream>>>(Wp1, Wp1t, 768, 256);
  cvt_tr<<<512, 256, 0, stream>>>(Wp2, Wp2t, 1024, 256);
  cvt_tr<<<512, 256, 0, stream>>>(Win, Wint, 256, 768);
  cvt_tr<<<256, 256, 0, stream>>>(Wout, Woutt, 256, 256);

  dim3 blk(32, 1, 1);
  // proj GEMMs (ReLU fused; output interleaved as [B, mod, 256] == [3B,256])
  gemm_wmma<<<dim3(4, 256), blk, 0, stream>>>(x0b, Wp0t, bp0, seqpre + 0,   nullptr, 512,  512,  768, 1);
  gemm_wmma<<<dim3(4, 256), blk, 0, stream>>>(x1b, Wp1t, bp1, seqpre + 256, nullptr, 768,  768,  768, 1);
  gemm_wmma<<<dim3(4, 256), blk, 0, stream>>>(x2b, Wp2t, bp2, seqpre + 512, nullptr, 1024, 1024, 768, 1);
  ln_kernel<<<24576 / 8, 256, 0, stream>>>(seqpre, seqb, g0, g1, g2, be0, be1, be2);
  // QKV GEMM [3B,256]@[256,768] -> f16
  gemm_wmma<<<dim3(12, 768), blk, 0, stream>>>(seqb, Wint, binv, nullptr, qkvb, 256, 256, 768, 0);
  attn_kernel<<<1024, 256, 0, stream>>>(qkvb, ob);
  // output GEMM [3B,256]@[256,256]
  gemm_wmma<<<dim3(4, 768), blk, 0, stream>>>(ob, Woutt, bout, oWp, nullptr, 256, 256, 256, 0);
  mean_kernel<<<8192, 256, 0, stream>>>(oWp, fusedf, fusedh);
  // similarity GEMM: fused @ fused^T -> raw scores staged in H (d_out)
  gemm_wmma<<<dim3(128, 256), blk, 0, stream>>>(fusedh, fusedh, nullptr, H, nullptr, 256, 256, 8192, 0);
  softmax_topk<<<8192, 256, 0, stream>>>(H, tvals, tidx);
  init_eye<<<8192, 256, 0, stream>>>(H);
  scatter_kernel<<<32, 256, 0, stream>>>(H, tvals, tidx);
  edge_weight<<<1024, 256, 0, stream>>>(fusedf, W1, b1, W2, b2, ew);
}
